// N2JNet_70574902608635
// MI455X (gfx1250) — matrix operand order repaired and checked by reading, not compiled, not run on hardware
//
#include <hip/hip_runtime.h>

// ============================================================================
// GraphNet (node MLP + global segment-sum MLP, L=20 meta-layers) on gfx1250.
//
// Strategy: one wave32 == one 16-row tile. All GEMMs are
// v_wmma_f32_16x16x32_f16 with K padded to 32/64 and N padded to 32.
// Weights are staged to LDS as f16, column-major ([n][k]) so each B fragment
// is two contiguous 16B DS reads per lane. Activations round-trip through LDS
// (f16) between GEMMs to realize the D-layout -> A-layout transpose.
// LayerNorm runs lane-per-row from an f32 LDS tile. segment_sum uses
// global_atomic_add_f32 (graphs live in L2: 8192*20*4 = 640 KB).
//
// d_out = [ x : N*20 f32 | u : 8192*20 f32 ]   d_ws = agg (8192*20 f32)
// ============================================================================

typedef __attribute__((ext_vector_type(16))) _Float16 v16h;
typedef __attribute__((ext_vector_type(8)))  _Float16 v8h;
typedef __attribute__((ext_vector_type(8)))  float    v8f;

#define NUM_GRAPHS      8192
#define LN_EPS          1e-5f
#define WAVES_PER_BLOCK 8
#define BLOCK_THREADS   256

// ---- WMMA fragment loaders ------------------------------------------------
// A (16x32 f16): lane holds row m=lane%16; halves 0-15 = K base..base+15,
// base = (lane<16 ? 0 : 16).
__device__ __forceinline__ v16h lds_loadA(const _Float16* p) {
  v16h r;
  ((v8h*)&r)[0] = *((const v8h*)(p));       // K base .. base+7
  ((v8h*)&r)[1] = *((const v8h*)(p + 8));   // K base+8 .. base+15
  return r;
}
// B (32x16 f16): lane holds col n=lane%16; halves 0-15 = K kbase..kbase+15,
// kbase = (lane<16 ? 0 : 16). Weights stored column-major -> contiguous.
__device__ __forceinline__ v16h lds_loadB(const _Float16* p) {
  v16h r;
  ((v8h*)&r)[0] = ((const v8h*)p)[0];
  ((v8h*)&r)[1] = ((const v8h*)p)[1];
  return r;
}

__device__ __forceinline__ v8f wmma_f16(v16h a, v16h b, v8f c) {
  return __builtin_amdgcn_wmma_f32_16x16x32_f16(false, a, false, b,
                                                (short)0, c, false, false);
}

// Stage W (Kact x 20 row-major f32) into LDS as f16 column-major [32][Kpad].
__device__ __forceinline__ void stage_wT(const float* __restrict__ W, int Kact,
                                         int Kpad, _Float16* wT, int tid,
                                         int nth) {
  const int total = 32 * Kpad;
  for (int idx = tid; idx < total; idx += nth) {
    const int n = idx / Kpad;
    const int k = idx - n * Kpad;
    float v = (n < 20 && k < Kact) ? W[k * 20 + n] : 0.0f;
    wT[idx] = (_Float16)v;
  }
}

// Per-lane column bias for N-tile nt (valid cols 0..19).
__device__ __forceinline__ float biasN(const float* b, int nt, int lane) {
  const int n = (lane & 15) + 16 * nt;
  return (n < 20) ? b[n] : 0.0f;
}

// ---- 3-layer MLP on one 16-row tile ---------------------------------------
// inbuf: [16][32*KCH] f16 row-major.  mid: [16][32] f16 scratch.
// outc : [16][32] f32 result (D layout stored row-major). No LN here.
template <int KCH>
__device__ __forceinline__ void mlp3_tile(
    const _Float16* inbuf, _Float16* mid, float* outc, const _Float16* wT1,
    const _Float16* wT2, const _Float16* wT3, float b1lo, float b1hi,
    float b2lo, float b2hi, float b3lo, float b3hi, int lane) {
  const int nh    = lane & 15;
  const int abase = (lane & 16) ? 16 : 0;
  const int bbase = (lane & 16) ? 16 : 0;
  const int mrow  = (lane & 16) ? 8 : 0;

  // GEMM1: [16 x 32*KCH] x [32*KCH x 32] -> relu -> mid (f16)
  v16h a[KCH];
#pragma unroll
  for (int c = 0; c < KCH; ++c)
    a[c] = lds_loadA(inbuf + nh * (32 * KCH) + c * 32 + abase);
#pragma unroll
  for (int nt = 0; nt < 2; ++nt) {
    v8f acc = {};
#pragma unroll
    for (int c = 0; c < KCH; ++c) {
      v16h b = lds_loadB(wT1 + (nt * 16 + nh) * (32 * KCH) + c * 32 + bbase);
      acc = wmma_f16(a[c], b, acc);
    }
    const float bias = nt ? b1hi : b1lo;
#pragma unroll
    for (int i = 0; i < 8; ++i) {
      float v = acc[i] + bias;
      mid[(mrow + i) * 32 + nt * 16 + nh] = (_Float16)(v > 0.0f ? v : 0.0f);
    }
  }

  // GEMM2: [16x32] x [32x32] -> relu -> mid (A preloaded, in-wave order safe)
  v16h a2 = lds_loadA(mid + nh * 32 + abase);
#pragma unroll
  for (int nt = 0; nt < 2; ++nt) {
    v16h b  = lds_loadB(wT2 + (nt * 16 + nh) * 32 + bbase);
    v8f acc = {};
    acc = wmma_f16(a2, b, acc);
    const float bias = nt ? b2hi : b2lo;
#pragma unroll
    for (int i = 0; i < 8; ++i) {
      float v = acc[i] + bias;
      mid[(mrow + i) * 32 + nt * 16 + nh] = (_Float16)(v > 0.0f ? v : 0.0f);
    }
  }

  // GEMM3: [16x32] x [32x32] + bias -> f32 outc (no relu)
  v16h a3 = lds_loadA(mid + nh * 32 + abase);
#pragma unroll
  for (int nt = 0; nt < 2; ++nt) {
    v16h b  = lds_loadB(wT3 + (nt * 16 + nh) * 32 + bbase);
    v8f acc = {};
    acc = wmma_f16(a3, b, acc);
    const float bias = nt ? b3hi : b3lo;
#pragma unroll
    for (int i = 0; i < 8; ++i)
      outc[(mrow + i) * 32 + nt * 16 + nh] = acc[i] + bias;
  }
}

// ---- node-init MLP: x_in (N x 10) -> x (N x 20) ---------------------------
__global__ void __launch_bounds__(BLOCK_THREADS)
node_init_kernel(const float* __restrict__ xin, float* __restrict__ xout,
                 const float* W1, const float* b1, const float* W2,
                 const float* b2, const float* W3, const float* b3,
                 const float* g, const float* be, int N) {
  __shared__ __align__(16) _Float16 s_w1[32 * 32];
  __shared__ __align__(16) _Float16 s_w2[32 * 32];
  __shared__ __align__(16) _Float16 s_w3[32 * 32];
  __shared__ __align__(16) _Float16 s_in[WAVES_PER_BLOCK][16 * 32];
  __shared__ __align__(16) _Float16 s_mid[WAVES_PER_BLOCK][16 * 32];
  __shared__ __align__(16) float    s_c[WAVES_PER_BLOCK][16 * 32];

  const int tid = threadIdx.x;
  stage_wT(W1, 10, 32, s_w1, tid, BLOCK_THREADS);
  stage_wT(W2, 20, 32, s_w2, tid, BLOCK_THREADS);
  stage_wT(W3, 20, 32, s_w3, tid, BLOCK_THREADS);
  __syncthreads();

  const int wave = tid >> 5, lane = tid & 31;
  const float b1v0 = biasN(b1, 0, lane), b1v1 = biasN(b1, 1, lane);
  const float b2v0 = biasN(b2, 0, lane), b2v1 = biasN(b2, 1, lane);
  const float b3v0 = biasN(b3, 0, lane), b3v1 = biasN(b3, 1, lane);
  _Float16* in  = s_in[wave];
  _Float16* mid = s_mid[wave];
  float*    cc  = s_c[wave];

  const int row = lane & 15, part = lane >> 4;
  const int ntiles = (N + 15) >> 4;
  const int slots  = gridDim.x * WAVES_PER_BLOCK;
  for (int tile = blockIdx.x * WAVES_PER_BLOCK + wave; tile < ntiles;
       tile += slots) {
    const int node = tile * 16 + row;
    _Float16* rp = in + row * 32;
    if (part == 0) {
      if (node < N) {
#pragma unroll
        for (int j = 0; j < 10; ++j) rp[j] = (_Float16)xin[node * 10 + j];
      } else {
        for (int j = 0; j < 10; ++j) rp[j] = (_Float16)0.0f;
      }
#pragma unroll
      for (int j = 10; j < 16; ++j) rp[j] = (_Float16)0.0f;
    } else {
#pragma unroll
      for (int j = 16; j < 32; ++j) rp[j] = (_Float16)0.0f;
    }

    mlp3_tile<1>(in, mid, cc, s_w1, s_w2, s_w3, b1v0, b1v1, b2v0, b2v1, b3v0,
                 b3v1, lane);

    if (part == 0 && node < N) {
      const float* cr = cc + row * 32;
      float s = 0.0f;
      for (int j = 0; j < 20; ++j) s += cr[j];
      const float mu = s * 0.05f;
      float var = 0.0f;
      for (int j = 0; j < 20; ++j) { float d = cr[j] - mu; var += d * d; }
      const float rstd = rsqrtf(var * 0.05f + LN_EPS);
      for (int j = 0; j < 20; ++j)
        xout[node * 20 + j] = (cr[j] - mu) * rstd * g[j] + be[j];
    }
  }
}

// ---- per-layer fused node kernel: NodeModel + GlobalPre + scatter-add -----
__global__ void __launch_bounds__(BLOCK_THREADS)
node_layer_kernel(float* __restrict__ x, const float* __restrict__ u,
                  const int* __restrict__ batch, float* __restrict__ agg,
                  const float* nW1, const float* nb1, const float* nW2,
                  const float* nb2, const float* nW3, const float* nb3,
                  const float* ng, const float* nbe, const float* pW1,
                  const float* pb1, const float* pW2, const float* pb2,
                  const float* pW3, const float* pb3, const float* pg,
                  const float* pbe, int N) {
  __shared__ __align__(16) _Float16 s_nw1[32 * 64];
  __shared__ __align__(16) _Float16 s_nw2[32 * 32];
  __shared__ __align__(16) _Float16 s_nw3[32 * 32];
  __shared__ __align__(16) _Float16 s_pw1[32 * 64];
  __shared__ __align__(16) _Float16 s_pw2[32 * 32];
  __shared__ __align__(16) _Float16 s_pw3[32 * 32];
  __shared__ __align__(16) _Float16 s_in[WAVES_PER_BLOCK][16 * 64];
  __shared__ __align__(16) _Float16 s_mid[WAVES_PER_BLOCK][16 * 32];
  __shared__ __align__(16) float    s_c[WAVES_PER_BLOCK][16 * 32];

  const int tid = threadIdx.x;
  stage_wT(nW1, 40, 64, s_nw1, tid, BLOCK_THREADS);
  stage_wT(nW2, 20, 32, s_nw2, tid, BLOCK_THREADS);
  stage_wT(nW3, 20, 32, s_nw3, tid, BLOCK_THREADS);
  stage_wT(pW1, 40, 64, s_pw1, tid, BLOCK_THREADS);
  stage_wT(pW2, 20, 32, s_pw2, tid, BLOCK_THREADS);
  stage_wT(pW3, 20, 32, s_pw3, tid, BLOCK_THREADS);
  __syncthreads();

  const int wave = tid >> 5, lane = tid & 31;
  const float nb1v0 = biasN(nb1, 0, lane), nb1v1 = biasN(nb1, 1, lane);
  const float nb2v0 = biasN(nb2, 0, lane), nb2v1 = biasN(nb2, 1, lane);
  const float nb3v0 = biasN(nb3, 0, lane), nb3v1 = biasN(nb3, 1, lane);
  const float pb1v0 = biasN(pb1, 0, lane), pb1v1 = biasN(pb1, 1, lane);
  const float pb2v0 = biasN(pb2, 0, lane), pb2v1 = biasN(pb2, 1, lane);
  const float pb3v0 = biasN(pb3, 0, lane), pb3v1 = biasN(pb3, 1, lane);
  _Float16* in  = s_in[wave];
  _Float16* mid = s_mid[wave];
  float*    cc  = s_c[wave];

  const int row = lane & 15, part = lane >> 4;
  const int ntiles = (N + 15) >> 4;
  const int slots  = gridDim.x * WAVES_PER_BLOCK;
  for (int tile = blockIdx.x * WAVES_PER_BLOCK + wave; tile < ntiles;
       tile += slots) {
    const int node = tile * 16 + row;
    _Float16* rp = in + row * 64;
    // Stage concat[x, u[batch]] as f16, K padded 40 -> 64.
    if (part == 0) {
      if (node < N) {
#pragma unroll
        for (int j = 0; j < 20; ++j) rp[j] = (_Float16)x[node * 20 + j];
      } else {
        for (int j = 0; j < 20; ++j) rp[j] = (_Float16)0.0f;
      }
#pragma unroll
      for (int j = 40; j < 52; ++j) rp[j] = (_Float16)0.0f;
    } else {
      if (node < N) {
        const int gid = batch[node];
#pragma unroll
        for (int j = 0; j < 20; ++j) rp[20 + j] = (_Float16)u[gid * 20 + j];
      } else {
        for (int j = 0; j < 20; ++j) rp[20 + j] = (_Float16)0.0f;
      }
#pragma unroll
      for (int j = 52; j < 64; ++j) rp[j] = (_Float16)0.0f;
    }

    // NodeModel MLP
    mlp3_tile<2>(in, mid, cc, s_nw1, s_nw2, s_nw3, nb1v0, nb1v1, nb2v0, nb2v1,
                 nb3v0, nb3v1, lane);

    // LayerNorm + residual -> x; refresh staged x-half for the pre-MLP.
    if (part == 0 && node < N) {
      const float* cr = cc + row * 32;
      float s = 0.0f;
      for (int j = 0; j < 20; ++j) s += cr[j];
      const float mu = s * 0.05f;
      float var = 0.0f;
      for (int j = 0; j < 20; ++j) { float d = cr[j] - mu; var += d * d; }
      const float rstd = rsqrtf(var * 0.05f + LN_EPS);
      for (int j = 0; j < 20; ++j) {
        const float xn =
            (cr[j] - mu) * rstd * ng[j] + nbe[j] + x[node * 20 + j];
        x[node * 20 + j] = xn;
        rp[j] = (_Float16)xn;
      }
    }

    // GlobalModel pre-aggregation MLP on concat[x_new, u[batch]].
    mlp3_tile<2>(in, mid, cc, s_pw1, s_pw2, s_pw3, pb1v0, pb1v1, pb2v0, pb2v1,
                 pb3v0, pb3v1, lane);

    // LayerNorm, then segment-sum via f32 global atomics (hits L2).
    if (part == 0 && node < N) {
      const float* cr = cc + row * 32;
      float s = 0.0f;
      for (int j = 0; j < 20; ++j) s += cr[j];
      const float mu = s * 0.05f;
      float var = 0.0f;
      for (int j = 0; j < 20; ++j) { float d = cr[j] - mu; var += d * d; }
      const float rstd = rsqrtf(var * 0.05f + LN_EPS);
      const int gid = batch[node];
      for (int j = 0; j < 20; ++j) {
        const float pv = (cr[j] - mu) * rstd * pg[j] + pbe[j];
        unsafeAtomicAdd(&agg[gid * 20 + j], pv);
      }
    }
  }
}

// ---- per-layer global update: u = LN(MLP(concat[agg,u])) + u; agg = 0 -----
__global__ void __launch_bounds__(BLOCK_THREADS)
global_update_kernel(float* __restrict__ u, float* __restrict__ agg,
                     const float* qW1, const float* qb1, const float* qW2,
                     const float* qb2, const float* qW3, const float* qb3,
                     const float* qg, const float* qbe) {
  __shared__ __align__(16) _Float16 s_w1[32 * 64];
  __shared__ __align__(16) _Float16 s_w2[32 * 32];
  __shared__ __align__(16) _Float16 s_w3[32 * 32];
  __shared__ __align__(16) _Float16 s_in[WAVES_PER_BLOCK][16 * 64];
  __shared__ __align__(16) _Float16 s_mid[WAVES_PER_BLOCK][16 * 32];
  __shared__ __align__(16) float    s_c[WAVES_PER_BLOCK][16 * 32];

  const int tid = threadIdx.x;
  stage_wT(qW1, 40, 64, s_w1, tid, BLOCK_THREADS);
  stage_wT(qW2, 20, 32, s_w2, tid, BLOCK_THREADS);
  stage_wT(qW3, 20, 32, s_w3, tid, BLOCK_THREADS);
  __syncthreads();

  const int wave = tid >> 5, lane = tid & 31;
  const float b1v0 = biasN(qb1, 0, lane), b1v1 = biasN(qb1, 1, lane);
  const float b2v0 = biasN(qb2, 0, lane), b2v1 = biasN(qb2, 1, lane);
  const float b3v0 = biasN(qb3, 0, lane), b3v1 = biasN(qb3, 1, lane);
  _Float16* in  = s_in[wave];
  _Float16* mid = s_mid[wave];
  float*    cc  = s_c[wave];

  const int row = lane & 15, part = lane >> 4;
  const int ntiles = NUM_GRAPHS / 16;  // 512, exact
  const int slots  = gridDim.x * WAVES_PER_BLOCK;
  for (int tile = blockIdx.x * WAVES_PER_BLOCK + wave; tile < ntiles;
       tile += slots) {
    const int gid = tile * 16 + row;
    _Float16* rp = in + row * 64;
    if (part == 0) {
#pragma unroll
      for (int j = 0; j < 20; ++j) rp[j] = (_Float16)agg[gid * 20 + j];
#pragma unroll
      for (int j = 40; j < 52; ++j) rp[j] = (_Float16)0.0f;
    } else {
#pragma unroll
      for (int j = 0; j < 20; ++j) rp[20 + j] = (_Float16)u[gid * 20 + j];
#pragma unroll
      for (int j = 52; j < 64; ++j) rp[j] = (_Float16)0.0f;
    }

    mlp3_tile<2>(in, mid, cc, s_w1, s_w2, s_w3, b1v0, b1v1, b2v0, b2v1, b3v0,
                 b3v1, lane);

    if (part == 0) {
      const float* cr = cc + row * 32;
      float s = 0.0f;
      for (int j = 0; j < 20; ++j) s += cr[j];
      const float mu = s * 0.05f;
      float var = 0.0f;
      for (int j = 0; j < 20; ++j) { float d = cr[j] - mu; var += d * d; }
      const float rstd = rsqrtf(var * 0.05f + LN_EPS);
      for (int j = 0; j < 20; ++j) {
        const float un =
            (cr[j] - mu) * rstd * qg[j] + qbe[j] + u[gid * 20 + j];
        u[gid * 20 + j]  = un;
        agg[gid * 20 + j] = 0.0f;  // ready for next layer's scatter-add
      }
    }
  }
}

// ---- host launcher --------------------------------------------------------
extern "C" void kernel_launch(void* const* d_in, const int* in_sizes, int n_in,
                              void* d_out, int out_size, void* d_ws,
                              size_t ws_size, hipStream_t stream) {
  const float* x_in  = (const float*)d_in[0];
  const int*   batch = (const int*)d_in[1];
  const float* ni_W1 = (const float*)d_in[2];
  const float* ni_b1 = (const float*)d_in[3];
  const float* ni_W2 = (const float*)d_in[4];
  const float* ni_b2 = (const float*)d_in[5];
  const float* ni_W3 = (const float*)d_in[6];
  const float* ni_b3 = (const float*)d_in[7];
  const float* ni_g  = (const float*)d_in[8];
  const float* ni_be = (const float*)d_in[9];
  const float* nm_W1 = (const float*)d_in[10];
  const float* nm_b1 = (const float*)d_in[11];
  const float* nm_W2 = (const float*)d_in[12];
  const float* nm_b2 = (const float*)d_in[13];
  const float* nm_W3 = (const float*)d_in[14];
  const float* nm_b3 = (const float*)d_in[15];
  const float* nm_g  = (const float*)d_in[16];
  const float* nm_be = (const float*)d_in[17];
  const float* gp_W1 = (const float*)d_in[18];
  const float* gp_b1 = (const float*)d_in[19];
  const float* gp_W2 = (const float*)d_in[20];
  const float* gp_b2 = (const float*)d_in[21];
  const float* gp_W3 = (const float*)d_in[22];
  const float* gp_b3 = (const float*)d_in[23];
  const float* gp_g  = (const float*)d_in[24];
  const float* gp_be = (const float*)d_in[25];
  const float* gq_W1 = (const float*)d_in[26];
  const float* gq_b1 = (const float*)d_in[27];
  const float* gq_W2 = (const float*)d_in[28];
  const float* gq_b2 = (const float*)d_in[29];
  const float* gq_W3 = (const float*)d_in[30];
  const float* gq_b3 = (const float*)d_in[31];
  const float* gq_g  = (const float*)d_in[32];
  const float* gq_be = (const float*)d_in[33];

  const int N  = in_sizes[1];  // batch has one entry per node
  float* x   = (float*)d_out;
  float* u   = x + (size_t)N * 20;
  float* agg = (float*)d_ws;  // 8192*20 f32 = 640 KB

  hipMemsetAsync(u, 0, (size_t)NUM_GRAPHS * 20 * sizeof(float), stream);
  hipMemsetAsync(agg, 0, (size_t)NUM_GRAPHS * 20 * sizeof(float), stream);

  const dim3 blk(BLOCK_THREADS);
  node_init_kernel<<<1024, blk, 0, stream>>>(x_in, x, ni_W1, ni_b1, ni_W2,
                                             ni_b2, ni_W3, ni_b3, ni_g, ni_be,
                                             N);
  const int L = 20;
  for (int l = 0; l < L; ++l) {
    node_layer_kernel<<<1024, blk, 0, stream>>>(
        x, u, batch, agg, nm_W1 + (size_t)l * 800, nm_b1 + l * 20,
        nm_W2 + (size_t)l * 400, nm_b2 + l * 20, nm_W3 + (size_t)l * 400,
        nm_b3 + l * 20, nm_g + l * 20, nm_be + l * 20,
        gp_W1 + (size_t)l * 800, gp_b1 + l * 20, gp_W2 + (size_t)l * 400,
        gp_b2 + l * 20, gp_W3 + (size_t)l * 400, gp_b3 + l * 20, gp_g + l * 20,
        gp_be + l * 20, N);
    global_update_kernel<<<64, blk, 0, stream>>>(
        u, agg, gq_W1 + (size_t)l * 800, gq_b1 + l * 20,
        gq_W2 + (size_t)l * 400, gq_b2 + l * 20, gq_W3 + (size_t)l * 400,
        gq_b3 + l * 20, gq_g + l * 20, gq_be + l * 20);
  }
}
